// LSTMEncoding_29076928594507
// MI455X (gfx1250) — matrix operand order repaired and checked by reading, not compiled, not run on hardware
//
#include <hip/hip_runtime.h>
#include <math.h>

typedef _Float16 half_t;
typedef __attribute__((ext_vector_type(16))) _Float16 v16h;
typedef __attribute__((ext_vector_type(8)))  _Float16 v8h;
typedef __attribute__((ext_vector_type(2)))  _Float16 v2h;
typedef __attribute__((ext_vector_type(8)))  float    v8f;
typedef __attribute__((ext_vector_type(4)))  float    v4f;
typedef __attribute__((ext_vector_type(2)))  float    v2f;

#define GLOVE_D 300
#define KP      320            // GLOVE_D padded to multiple of 32
#define DMODEL  1024
#define BATCH   64
#define SEQ     64
#define NTOK    (BATCH * SEQ)  // 4096
#define G4      (4 * DMODEL)   // 4096

// ---------- WMMA operand loaders (CDNA5 16-bit layouts, CDNA5 ISA) ----------

// A matrix 16x32 f16: lane m = lane%16; lanes 0-15 hold K = kk+0..7 and kk+16..23,
// lanes 16-31 hold K = kk+8..15 and kk+24..31. Two contiguous 16B chunks per lane.
__device__ __forceinline__ v16h load_a16(const half_t* rowPtr, int kk, int lane) {
    const int ko = (lane >> 4) * 8;
    v8h lo = *(const v8h*)(rowPtr + kk + ko);
    v8h hi = *(const v8h*)(rowPtr + kk + 16 + ko);
    v16h a;
#pragma unroll
    for (int i = 0; i < 8; ++i) { a[i] = lo[i]; a[8 + i] = hi[i]; }
    return a;
}

// B matrix 32x16 f16 (column-major per column): lane n = lane%16;
// lanes 0-15 hold K = kk+0..7 and kk+16..23, lanes 16-31 hold K = kk+8..15 and kk+24..31.
__device__ __forceinline__ v16h load_b16(const half_t* colPtr, int kk, int lane) {
    const int ko = (lane >> 4) * 8;
    v8h lo = *(const v8h*)(colPtr + kk + ko);
    v8h hi = *(const v8h*)(colPtr + kk + 16 + ko);
    v16h b;
#pragma unroll
    for (int i = 0; i < 8; ++i) { b[i] = lo[i]; b[8 + i] = hi[i]; }
    return b;
}

__device__ __forceinline__ v8f wmma_f16(v16h a, v16h b, v8f c) {
    return __builtin_amdgcn_wmma_f32_16x16x32_f16(false, a, false, b, (short)0, c,
                                                  false, false);
}

__device__ __forceinline__ float sigmoidf_(float x) {
    return 1.0f / (1.0f + __expf(-x));   // large |x| saturates cleanly (inf -> 0)
}

__device__ __forceinline__ float tanhf_(float x) {
    float e = __expf(-2.0f * fabsf(x));  // e in (0,1], no overflow
    float r = (1.0f - e) / (1.0f + e);
    return (x >= 0.0f) ? r : -r;
}

// ---------- pre-pass kernels (branch-free f32->f16 staging) ----------

// Gather glove rows for every token, convert to f16, zero-pad 300 -> 320.
// One thread per (row, k-pair); 300 is even so a pair never straddles the pad.
__global__ void gather_pad_kernel(const int* __restrict__ seqs,
                                  const float* __restrict__ glove,
                                  half_t* __restrict__ emb16) {
    int idx = blockIdx.x * blockDim.x + threadIdx.x;   // 4096 * 160 pairs
    if (idx >= NTOK * (KP / 2)) return;
    int row = idx / (KP / 2);
    int k   = (idx - row * (KP / 2)) * 2;
    v2h o;
    if (k < GLOVE_D) {
        v2f g = *(const v2f*)(glove + (size_t)seqs[row] * GLOVE_D + k);
        o[0] = (half_t)g[0];
        o[1] = (half_t)g[1];
    } else {
        o[0] = (half_t)0.0f;
        o[1] = (half_t)0.0f;
    }
    *(v2h*)(emb16 + (size_t)row * KP + k) = o;
}

// W_emb [1024 x 300] -> f16 padded [1024 x 320].
__global__ void wemb_pad_kernel(const float* __restrict__ wemb,
                                half_t* __restrict__ wemb16) {
    int idx = blockIdx.x * blockDim.x + threadIdx.x;   // 1024 * 160 pairs
    if (idx >= DMODEL * (KP / 2)) return;
    int row = idx / (KP / 2);
    int k   = (idx - row * (KP / 2)) * 2;
    v2h o;
    if (k < GLOVE_D) {
        v2f g = *(const v2f*)(wemb + (size_t)row * GLOVE_D + k);
        o[0] = (half_t)g[0];
        o[1] = (half_t)g[1];
    } else {
        o[0] = (half_t)0.0f;
        o[1] = (half_t)0.0f;
    }
    *(v2h*)(wemb16 + (size_t)row * KP + k) = o;
}

// W_ih / W_hh f32 -> f16, 2 elements per thread (packed cvt).
__global__ void cvt_weights_f16(const float* __restrict__ wih,
                                const float* __restrict__ whh,
                                half_t* __restrict__ wih16,
                                half_t* __restrict__ whh16, int npairs) {
    int i = blockIdx.x * blockDim.x + threadIdx.x;
    if (i < npairs) {
        v2f a = *(const v2f*)(wih + (size_t)i * 2);
        v2f b = *(const v2f*)(whh + (size_t)i * 2);
        v2h oa, ob;
        oa[0] = (half_t)a[0]; oa[1] = (half_t)a[1];
        ob[0] = (half_t)b[0]; ob[1] = (half_t)b[1];
        *(v2h*)(wih16 + (size_t)i * 2) = oa;
        *(v2h*)(whh16 + (size_t)i * 2) = ob;
    }
}

__global__ void init_state(half_t* __restrict__ h16, float* __restrict__ c, int n) {
    int i = blockIdx.x * blockDim.x + threadIdx.x;
    if (i < n) {
        h16[i] = (half_t)0.0f;
        c[i]   = 0.0f;
    }
}

// ---------- stage 1: embed projection (branch-free WMMA) ----------
// x[token, col] = sum_k emb16[token, k] * wemb16[col, k] + b_emb[col]
__global__ void embed_kernel(const half_t* __restrict__ emb16,
                             const half_t* __restrict__ wemb16,
                             const float* __restrict__ bemb,
                             half_t* __restrict__ xf16) {
    const int lane  = threadIdx.x & 31;
    const int wave  = threadIdx.x >> 5;
    const int nTile = blockIdx.x;               // 0..63   (1024 cols)
    const int mTile = blockIdx.y * 4 + wave;    // 0..255  (4096 tokens)
    const int hi    = lane >> 4;
    const int n     = lane & 15;

    const half_t* arow = emb16 + (size_t)(mTile * 16 + n) * KP;
    const int col = nTile * 16 + n;
    const half_t* bcol = wemb16 + (size_t)col * KP;

    v8f acc = {};
#pragma unroll 2
    for (int kk = 0; kk < KP; kk += 32) {       // 10 clean k-steps
        v16h a = load_a16(arow, kk, lane);
        v16h b = load_b16(bcol, kk, lane);
        acc = wmma_f16(a, b, acc);
    }

    const float bias = bemb[col];
#pragma unroll
    for (int r = 0; r < 8; ++r) {
        int row = mTile * 16 + r + hi * 8;
        xf16[(size_t)row * DMODEL + col] = (half_t)(acc[r] + bias);
    }
}

// ---------- stage 2: input projection for all timesteps ----------
// x_proj[t, b, col] = sum_k x[token, k] * W_ih[col, k] + b_ih[col] + b_hh[col]
// Each wave computes a 2x2 quad of 16x16 tiles: 2 A + 2 B loads feed 4 WMMAs.
__global__ void proj_kernel(const half_t* __restrict__ xf16,
                            const half_t* __restrict__ wih16,
                            const float* __restrict__ bih,
                            const float* __restrict__ bhh,
                            float* __restrict__ xproj) {
    const int lane  = threadIdx.x & 31;
    const int wave  = threadIdx.x >> 5;
    const int nPair = blockIdx.x;               // 0..127 (pairs of col tiles)
    const int mPair = blockIdx.y * 4 + wave;    // 0..127 (pairs of token tiles)
    const int hi    = lane >> 4;
    const int n     = lane & 15;

    const int m0   = mPair * 32;
    const int col0 = nPair * 32;

    const half_t* arow0 = xf16 + (size_t)(m0 + n) * DMODEL;
    const half_t* arow1 = xf16 + (size_t)(m0 + 16 + n) * DMODEL;
    const half_t* bcol0 = wih16 + (size_t)(col0 + n) * DMODEL;
    const half_t* bcol1 = wih16 + (size_t)(col0 + 16 + n) * DMODEL;

    v8f acc00 = {}, acc01 = {}, acc10 = {}, acc11 = {};
#pragma unroll 2
    for (int kk = 0; kk < DMODEL; kk += 32) {
        v16h a0 = load_a16(arow0, kk, lane);
        v16h a1 = load_a16(arow1, kk, lane);
        v16h b0 = load_b16(bcol0, kk, lane);
        v16h b1 = load_b16(bcol1, kk, lane);
        acc00 = wmma_f16(a0, b0, acc00);
        acc01 = wmma_f16(a0, b1, acc01);
        acc10 = wmma_f16(a1, b0, acc10);
        acc11 = wmma_f16(a1, b1, acc11);
    }

    const float bias0 = bih[col0 + n] + bhh[col0 + n];
    const float bias1 = bih[col0 + 16 + n] + bhh[col0 + 16 + n];
#pragma unroll
    for (int r = 0; r < 8; ++r) {
        const int mOff = r + hi * 8;
        // tile (0, *): token row m0 + mOff
        {
            int row = m0 + mOff;
            int bb = row >> 6, tt = row & 63;
            size_t base = ((size_t)tt * BATCH + bb) * G4;
            xproj[base + col0 + n]      = acc00[r] + bias0;
            xproj[base + col0 + 16 + n] = acc01[r] + bias1;
        }
        // tile (1, *): token row m0 + 16 + mOff
        {
            int row = m0 + 16 + mOff;
            int bb = row >> 6, tt = row & 63;
            size_t base = ((size_t)tt * BATCH + bb) * G4;
            xproj[base + col0 + n]      = acc10[r] + bias0;
            xproj[base + col0 + 16 + n] = acc11[r] + bias1;
        }
    }
}

// ---------- stage 3: one LSTM timestep ----------
// 256 threads/block = 8 waves. Waves are (mTile 0..3) x (kHalf 0..1); each
// computes the i/f/g/o 16x16 tiles over half of K (16 k-steps). kHalf=1 waves
// spill accumulators to LDS; kHalf=0 waves reduce, apply gates, write h/c/out.
// h double-buffered across steps (f16).
__global__ void lstm_step_kernel(const half_t* __restrict__ hin,
                                 const half_t* __restrict__ whh16,
                                 const float* __restrict__ xproj_t,
                                 float* __restrict__ c,
                                 half_t* __restrict__ hout,
                                 float* __restrict__ out, int t) {
    __shared__ float red[4 * 4 * 32 * 8];   // [mt][gate][lane][r] = 16 KB

    const int lane  = threadIdx.x & 31;
    const int wave  = threadIdx.x >> 5;
    const int mt    = wave & 3;           // batch tile 0..3
    const int kh    = wave >> 2;          // K half 0..1
    const int nTile = blockIdx.x;         // 0..63 (1024 d cols)
    const int hi    = lane >> 4;
    const int n     = lane & 15;

    const half_t* arow = hin + (size_t)(mt * 16 + n) * DMODEL;
    const int colD = nTile * 16 + n;
    const half_t* bi = whh16 + (size_t)(colD)        * DMODEL;
    const half_t* bf = whh16 + (size_t)(colD + 1024) * DMODEL;
    const half_t* bg = whh16 + (size_t)(colD + 2048) * DMODEL;
    const half_t* bo = whh16 + (size_t)(colD + 3072) * DMODEL;

    const int k0 = kh * (DMODEL / 2);
    v8f ai = {}, af = {}, ag = {}, ao = {};
#pragma unroll 2
    for (int kk = k0; kk < k0 + DMODEL / 2; kk += 32) {   // 16 k-steps
        v16h a  = load_a16(arow, kk, lane);
        v16h vi = load_b16(bi, kk, lane);
        v16h vf = load_b16(bf, kk, lane);
        v16h vg = load_b16(bg, kk, lane);
        v16h vo = load_b16(bo, kk, lane);
        ai = wmma_f16(a, vi, ai);
        af = wmma_f16(a, vf, af);
        ag = wmma_f16(a, vg, ag);
        ao = wmma_f16(a, vo, ao);
    }

    if (kh == 1) {
        // spill: red[((mt*4+g)*32 + lane)*8 + r], two b128 stores per gate
        v8f* accs[4] = { &ai, &af, &ag, &ao };
#pragma unroll
        for (int g = 0; g < 4; ++g) {
            float* p = &red[(((mt * 4 + g) * 32) + lane) * 8];
            v8f a = *accs[g];
            *(v4f*)(p)     = (v4f){a[0], a[1], a[2], a[3]};
            *(v4f*)(p + 4) = (v4f){a[4], a[5], a[6], a[7]};
        }
    }
    __syncthreads();
    if (kh == 0) {
        v8f* accs[4] = { &ai, &af, &ag, &ao };
#pragma unroll
        for (int g = 0; g < 4; ++g) {
            const float* p = &red[(((mt * 4 + g) * 32) + lane) * 8];
            v4f lo = *(const v4f*)(p);
            v4f hi4 = *(const v4f*)(p + 4);
            v8f& a = *accs[g];
            a[0] += lo[0];  a[1] += lo[1];  a[2] += lo[2];  a[3] += lo[3];
            a[4] += hi4[0]; a[5] += hi4[1]; a[6] += hi4[2]; a[7] += hi4[3];
        }
#pragma unroll
        for (int r = 0; r < 8; ++r) {
            const int b = mt * 16 + r + hi * 8;   // batch row
            const size_t xb = (size_t)b * G4;
            float iv = sigmoidf_(ai[r] + xproj_t[xb + colD]);
            float fv = sigmoidf_(af[r] + xproj_t[xb + 1024 + colD]);
            float gv = tanhf_   (ag[r] + xproj_t[xb + 2048 + colD]);
            float ov = sigmoidf_(ao[r] + xproj_t[xb + 3072 + colD]);
            const size_t idx = (size_t)b * DMODEL + colD;
            float cN = fv * c[idx] + iv * gv;
            c[idx] = cN;
            float hN = ov * tanhf_(cN);
            hout[idx] = (half_t)hN;
            out[((size_t)b * SEQ + t) * DMODEL + colD] = hN;
        }
    }
}

// ---------- host ----------

extern "C" void kernel_launch(void* const* d_in, const int* in_sizes, int n_in,
                              void* d_out, int out_size, void* d_ws, size_t ws_size,
                              hipStream_t stream) {
    const int*   seqs  = (const int*)d_in[0];
    const float* glove = (const float*)d_in[1];
    const float* wemb  = (const float*)d_in[2];
    const float* bemb  = (const float*)d_in[3];
    const float* wih   = (const float*)d_in[4];
    const float* whh   = (const float*)d_in[5];
    const float* bih   = (const float*)d_in[6];
    const float* bhh   = (const float*)d_in[7];
    float* out = (float*)d_out;

    char* ws = (char*)d_ws;
    size_t off = 0;
    auto take = [&](size_t bytes) {
        void* p = ws + off;
        off += (bytes + 255) & ~(size_t)255;
        return p;
    };
    half_t* wih16  = (half_t*)take((size_t)G4 * DMODEL * sizeof(half_t));    // 8 MB
    half_t* whh16  = (half_t*)take((size_t)G4 * DMODEL * sizeof(half_t));    // 8 MB
    half_t* emb16  = (half_t*)take((size_t)NTOK * KP * sizeof(half_t));      // 2.5 MB
    half_t* wemb16 = (half_t*)take((size_t)DMODEL * KP * sizeof(half_t));    // 640 KB
    half_t* xf16   = (half_t*)take((size_t)NTOK * DMODEL * sizeof(half_t));  // 8 MB
    float*  xproj  = (float*)take((size_t)SEQ * BATCH * G4 * sizeof(float)); // 64 MB
    half_t* hA     = (half_t*)take((size_t)BATCH * DMODEL * sizeof(half_t));
    half_t* hB     = (half_t*)take((size_t)BATCH * DMODEL * sizeof(half_t));
    float*  cbuf   = (float*)take((size_t)BATCH * DMODEL * sizeof(float));

    // pre-pass staging
    {
        int np = NTOK * (KP / 2);
        gather_pad_kernel<<<(np + 255) / 256, 256, 0, stream>>>(seqs, glove, emb16);
    }
    {
        int np = DMODEL * (KP / 2);
        wemb_pad_kernel<<<(np + 255) / 256, 256, 0, stream>>>(wemb, wemb16);
    }
    {
        int np = (G4 * DMODEL) / 2;  // 2M pairs per weight matrix
        cvt_weights_f16<<<(np + 255) / 256, 256, 0, stream>>>(wih, whh, wih16, whh16, np);
    }
    {
        int ns = BATCH * DMODEL;
        init_state<<<(ns + 255) / 256, 256, 0, stream>>>(hA, cbuf, ns);
    }

    // stage 1: 64 col-tiles x (64 groups of 4 m-tiles), 4 waves/block
    embed_kernel<<<dim3(64, 64), 128, 0, stream>>>(emb16, wemb16, bemb, xf16);

    // stage 2: 128 col-pairs x (32 groups of 4 m-pairs), 2x2 tiles per wave
    proj_kernel<<<dim3(128, 32), 128, 0, stream>>>(xf16, wih16, bih, bhh, xproj);

    // stage 3: 64 sequential steps, h double-buffered, K split across wave pairs
    for (int t = 0; t < SEQ; ++t) {
        half_t* hin  = (t & 1) ? hB : hA;
        half_t* hout = (t & 1) ? hA : hB;
        lstm_step_kernel<<<64, 256, 0, stream>>>(
            hin, whh16, xproj + (size_t)t * BATCH * G4, cbuf, hout, out, t);
    }
}
